// DAWN_82901458747749
// MI455X (gfx1250) — matrix operand order repaired and checked by reading, not compiled, not run on hardware
//
#include <hip/hip_runtime.h>
#include <math.h>

// ---------------- model constants ----------------
#define Vv   32000
#define Dm   512
#define Lm   4
#define Hm   8
#define DHm  64
#define NRm  8
#define NCm  8
#define MIDm 32
#define NNm  64
#define Km   8
#define Bm   2
#define Sm   2048
#define Mrows (Bm * Sm)          // 4096
#define NCOMBm (NRm * NCm)       // 64

typedef __bf16 v16bf __attribute__((ext_vector_type(16)));
typedef __bf16 v8bf  __attribute__((ext_vector_type(8)));
typedef float  v8f   __attribute__((ext_vector_type(8)));
typedef unsigned int u32x4 __attribute__((ext_vector_type(4)));
typedef unsigned int u32x8 __attribute__((ext_vector_type(8)));

static __device__ __forceinline__ v8f wmma_bf16(v16bf a, v16bf b, v8f c) {
  return __builtin_amdgcn_wmma_f32_16x16x32_bf16(false, a, false, b, (short)0, c, false, false);
}

// Build a 16xbf16 fragment from two 16-byte aligned 8-element groups.
// Lowers to 2x b128 loads (global or ds depending on address space).
static __device__ __forceinline__ v16bf ld_frag2(const __bf16* p0, const __bf16* p1) {
  u32x4 a = *(const u32x4*)p0;
  u32x4 b = *(const u32x4*)p1;
  u32x8 c = __builtin_shufflevector(a, b, 0, 1, 2, 3, 4, 5, 6, 7);
  return __builtin_bit_cast(v16bf, c);
}

// ---------------- utility kernels ----------------
__global__ __launch_bounds__(256) void cvt_bf16_kernel(const float* __restrict__ in,
                                                       __bf16* __restrict__ out, long long n) {
  long long i = (long long)blockIdx.x * blockDim.x + threadIdx.x;
  if (i < n) out[i] = (__bf16)in[i];
}

// token_emb (V,D) f32 -> (D,V) bf16 for the tied head GEMM
__global__ __launch_bounds__(256) void transpose_cvt_kernel(const float* __restrict__ te,
                                                            __bf16* __restrict__ out) {
  long long i = (long long)blockIdx.x * blockDim.x + threadIdx.x;
  long long n = (long long)Vv * Dm;
  if (i < n) {
    int k = (int)(i / Vv);
    int t = (int)(i % Vv);
    out[i] = (__bf16)te[(long long)t * Dm + k];
  }
}

// generic bf16 transpose: in (R x C, row stride ldin) -> out (C x R, row stride R)
__global__ __launch_bounds__(256) void transpose_bf16_kernel(const __bf16* __restrict__ in,
                                                             int ldin,
                                                             __bf16* __restrict__ out,
                                                             int R, int C) {
  __shared__ __bf16 tile[32][34];
  int bx = blockIdx.x * 32, by = blockIdx.y * 32;
  int tx = threadIdx.x & 31, ty = threadIdx.x >> 5;  // 32 x 8
  #pragma unroll
  for (int i = ty; i < 32; i += 8)
    tile[i][tx] = in[(size_t)(by + i) * ldin + bx + tx];
  __syncthreads();
  #pragma unroll
  for (int i = ty; i < 32; i += 8)
    out[(size_t)(bx + i) * R + by + tx] = tile[tx][i];
}

__global__ __launch_bounds__(128) void embed_kernel(const int* __restrict__ tokens,
                                                    const float* __restrict__ te,
                                                    const float* __restrict__ pe,
                                                    float* __restrict__ x) {
  int row = blockIdx.x;              // b*S + s
  int tok = tokens[row];
  int s   = row % Sm;
  for (int d = threadIdx.x; d < Dm; d += 128)
    x[(size_t)row * Dm + d] = te[(size_t)tok * Dm + d] + pe[(size_t)s * Dm + d];
}

// layernorm, writes bf16 (strided dest)
__global__ __launch_bounds__(128) void ln_kernel(const float* __restrict__ x,
                                                 const float* __restrict__ g,
                                                 const float* __restrict__ b,
                                                 __bf16* __restrict__ out, int ldo) {
  __shared__ float red[128];
  int row = blockIdx.x, t = threadIdx.x;
  const float* xr = x + (size_t)row * Dm;
  float v[4], s = 0.f;
  #pragma unroll
  for (int u = 0; u < 4; ++u) { v[u] = xr[t + u * 128]; s += v[u]; }
  red[t] = s; __syncthreads();
  for (int o = 64; o > 0; o >>= 1) { if (t < o) red[t] += red[t + o]; __syncthreads(); }
  float mean = red[0] * (1.f / Dm);
  __syncthreads();
  float var = 0.f;
  #pragma unroll
  for (int u = 0; u < 4; ++u) { float d = v[u] - mean; var += d * d; }
  red[t] = var; __syncthreads();
  for (int o = 64; o > 0; o >>= 1) { if (t < o) red[t] += red[t + o]; __syncthreads(); }
  float inv = rsqrtf(red[0] * (1.f / Dm) + 1e-5f);
  #pragma unroll
  for (int u = 0; u < 4; ++u) {
    int d = t + u * 128;
    out[(size_t)row * ldo + d] = (__bf16)((v[u] - mean) * inv * g[d] + b[d]);
  }
}

// basis_emb[comb][d] = normalize(0.5*(mean_k rb[i,d,k] + mean_m cb[j,m,d]))
__global__ __launch_bounds__(128) void basis_kernel(const float* __restrict__ rb,
                                                    const float* __restrict__ cb,
                                                    float* __restrict__ basis_emb) {
  __shared__ float red[128];
  int comb = blockIdx.x, i = comb >> 3, j = comb & 7, t = threadIdx.x;
  float vals[4], ss = 0.f;
  #pragma unroll
  for (int u = 0; u < 4; ++u) {
    int d = t * 4 + u;
    float rm = 0.f;
    for (int k2 = 0; k2 < MIDm; ++k2) rm += rb[((size_t)i * Dm + d) * MIDm + k2];
    float cm = 0.f;
    for (int m2 = 0; m2 < MIDm; ++m2) cm += cb[((size_t)j * MIDm + m2) * Dm + d];
    float e = 0.5f * (rm * (1.f / MIDm) + cm * (1.f / MIDm));
    vals[u] = e; ss += e * e;
  }
  red[t] = ss; __syncthreads();
  for (int o = 64; o > 0; o >>= 1) { if (t < o) red[t] += red[t + o]; __syncthreads(); }
  float inv = rsqrtf(red[0]);
  #pragma unroll
  for (int u = 0; u < 4; ++u) basis_emb[(size_t)comb * Dm + t * 4 + u] = vals[u] * inv;
}

// row_sum (8,32), col_sum_j (8,32), col_sum_i (8,512)
__global__ __launch_bounds__(256) void sums_kernel(const float* __restrict__ rb,
                                                   const float* __restrict__ cb,
                                                   float* __restrict__ row_sum,
                                                   float* __restrict__ col_sum_j,
                                                   float* __restrict__ col_sum_i) {
  int n = blockIdx.x, t = threadIdx.x;
  if (t < MIDm) {
    float s = 0.f;
    for (int d = 0; d < Dm; ++d) s += rb[((size_t)n * Dm + d) * MIDm + t];
    row_sum[n * MIDm + t] = s;
    float s2 = 0.f;
    for (int d = 0; d < Dm; ++d) s2 += cb[((size_t)n * MIDm + t) * Dm + d];
    col_sum_j[n * MIDm + t] = s2;
  }
  for (int d = t; d < Dm; d += 256) {
    float s = 0.f;
    for (int m = 0; m < MIDm; ++m) s += cb[((size_t)n * MIDm + m) * Dm + d];
    col_sum_i[n * Dm + d] = s;
  }
}

__global__ __launch_bounds__(64) void recsm_kernel(const float* __restrict__ recipe,
                                                   float* __restrict__ rec_sm) {
  __shared__ float red[64];
  int nrow = blockIdx.x, t = threadIdx.x;
  float v = recipe[(size_t)nrow * NCOMBm + t];
  red[t] = v; __syncthreads();
  for (int o = 32; o > 0; o >>= 1) { if (t < o) red[t] = fmaxf(red[t], red[t + o]); __syncthreads(); }
  float mx = red[0]; __syncthreads();
  float e = expf(v - mx);
  red[t] = e; __syncthreads();
  for (int o = 32; o > 0; o >>= 1) { if (t < o) red[t] += red[t + o]; __syncthreads(); }
  rec_sm[(size_t)nrow * NCOMBm + t] = e / red[0];
}

// neuron_embT[d][n] = sum_c rec_sm[n][c]*basis_emb[c][d], stored bf16 (D x NN)
__global__ __launch_bounds__(256) void neuron_kernel(const float* __restrict__ rec_sm,
                                                     const float* __restrict__ basis_emb,
                                                     __bf16* __restrict__ neuronT) {
  int n = blockIdx.x;
  for (int d = threadIdx.x; d < Dm; d += 256) {
    float s = 0.f;
    for (int c = 0; c < NCOMBm; ++c) s += rec_sm[n * NCOMBm + c] * basis_emb[(size_t)c * Dm + d];
    neuronT[(size_t)d * NNm + n] = (__bf16)s;
  }
}

// top-k routing + recipe mixing + qs/ks2/transform
__global__ __launch_bounds__(64) void route_kernel(const float* __restrict__ scores,
                                                   const float* __restrict__ rec_sm,
                                                   const float* __restrict__ row_sum,
                                                   const float* __restrict__ col_sum_j,
                                                   const float* __restrict__ col_sum_i,
                                                   __bf16* __restrict__ qs_bf,
                                                   __bf16* __restrict__ ks_bf,
                                                   float* __restrict__ transform) {
  __shared__ float tmp[64], red[64], rnorm[64], roww[8], colw[8], topw[8];
  __shared__ int redi[64], topi[8];
  int row = blockIdx.x, t = threadIdx.x;
  tmp[t] = scores[(size_t)row * NNm + t];
  __syncthreads();
  for (int k2 = 0; k2 < Km; ++k2) {
    red[t] = tmp[t]; redi[t] = t; __syncthreads();
    for (int o = 32; o > 0; o >>= 1) {
      if (t < o && red[t + o] > red[t]) { red[t] = red[t + o]; redi[t] = redi[t + o]; }
      __syncthreads();
    }
    if (t == 0) { topi[k2] = redi[0]; topw[k2] = red[0]; }
    __syncthreads();
    if (t == topi[k2]) tmp[t] = -3.0e38f;
    __syncthreads();
  }
  if (t == 0) {
    float mx = topw[0];
    for (int k2 = 1; k2 < Km; ++k2) mx = fmaxf(mx, topw[k2]);
    float s = 0.f;
    for (int k2 = 0; k2 < Km; ++k2) { topw[k2] = expf(topw[k2] - mx); s += topw[k2]; }
    for (int k2 = 0; k2 < Km; ++k2) topw[k2] /= s;
  }
  __syncthreads();
  float tr = 0.f;
  for (int k2 = 0; k2 < Km; ++k2) tr += topw[k2] * rec_sm[topi[k2] * NCOMBm + t];
  red[t] = tr; __syncthreads();
  for (int o = 32; o > 0; o >>= 1) { if (t < o) red[t] = fmaxf(red[t], red[t + o]); __syncthreads(); }
  float mx = red[0]; __syncthreads();
  float e = expf(tr - mx);
  red[t] = e; __syncthreads();
  for (int o = 32; o > 0; o >>= 1) { if (t < o) red[t] += red[t + o]; __syncthreads(); }
  rnorm[t] = e / red[0];
  __syncthreads();
  if (t < 8) {
    float s = 0.f;
    for (int c = 0; c < 8; ++c) s += rnorm[t * 8 + c];
    roww[t] = s;
  } else if (t < 16) {
    int c = t - 8; float s = 0.f;
    for (int r = 0; r < 8; ++r) s += rnorm[r * 8 + c];
    colw[c] = s;
  }
  __syncthreads();
  if (t < 32) {
    float s = 0.f;
    for (int r = 0; r < 8; ++r) s += roww[r] * row_sum[r * MIDm + t];
    qs_bf[(size_t)row * MIDm + t] = (__bf16)s;
  } else {
    int m = t - 32; float s = 0.f;
    for (int c = 0; c < 8; ++c) s += colw[c] * col_sum_j[c * MIDm + m];
    ks_bf[(size_t)row * MIDm + m] = (__bf16)s;
  }
  for (int d = t; d < Dm; d += 64) {
    float s = 0.f;
    for (int c = 0; c < 8; ++c) s += colw[c] * col_sum_i[(size_t)c * Dm + d];
    transform[(size_t)row * Dm + d] = s;
  }
}

// ---------------- generic WMMA GEMM ----------------
// C[M,N] = A[M,K]·B[K,N] (+bias)(+residual); 64x64 tile / block, 4 waves, K-step 32.
// A fragments load straight from global (2x b128 per lane); B staged in LDS
// K-transposed so every B fragment is 2x ds_load_b128.
__global__ __launch_bounds__(128) void wmma_gemm_kernel(
    const __bf16* __restrict__ A, int lda,
    const __bf16* __restrict__ Bmat, int ldb,
    const float* __restrict__ bias,
    const float* __restrict__ residual, int ldr,
    float* __restrict__ Cf, int ldcf,
    __bf16* __restrict__ Cb, int ldcb,
    int Kd) {
  __shared__ __bf16 Bs[64][40];   // [col][k], 80B row stride (16B aligned)
  int t = threadIdx.x, lane = t & 31, w = t >> 5;
  int half = lane >> 4, lq = lane & 15;
  int row0 = blockIdx.y * 64, col0 = blockIdx.x * 64;
  const __bf16* arow = A + (size_t)(row0 + 16 * w + lq) * lda;
  v8f acc[4];
  #pragma unroll
  for (int nb = 0; nb < 4; ++nb)
    #pragma unroll
    for (int r = 0; r < 8; ++r) acc[nb][r] = 0.f;

  for (int k0 = 0; k0 < Kd; k0 += 32) {
    // stage B tile 32x64 -> Bs[col][k] (transposed)
    #pragma unroll
    for (int it = 0; it < 2; ++it) {
      int g = t + 128 * it;           // 0..255
      int r = g >> 3, cg = g & 7;     // k-row, 8-col group
      u32x4 val = *(const u32x4*)&Bmat[(size_t)(k0 + r) * ldb + col0 + cg * 8];
      v8bf e = __builtin_bit_cast(v8bf, val);
      #pragma unroll
      for (int j = 0; j < 8; ++j) Bs[cg * 8 + j][r] = e[j];
    }
    if (k0 + 32 < Kd)
      __builtin_prefetch(&Bmat[(size_t)(k0 + 32 + (t >> 2)) * ldb + col0], 0, 1);
    __syncthreads();
    v16bf af = ld_frag2(arow + k0 + half * 8, arow + k0 + 16 + half * 8);
    #pragma unroll
    for (int nb = 0; nb < 4; ++nb) {
      v16bf bv = ld_frag2(&Bs[nb * 16 + lq][half * 8], &Bs[nb * 16 + lq][16 + half * 8]);
      acc[nb] = wmma_bf16(af, bv, acc[nb]);
    }
    __syncthreads();
  }
  #pragma unroll
  for (int nb = 0; nb < 4; ++nb)
    #pragma unroll
    for (int r = 0; r < 8; ++r) {
      int grow = row0 + 16 * w + 8 * half + r;
      int gcol = col0 + nb * 16 + lq;
      float v2 = acc[nb][r];
      if (bias) v2 += bias[gcol];
      if (residual) v2 += residual[(size_t)grow * ldr + gcol];
      if (Cf) Cf[(size_t)grow * ldcf + gcol] = v2;
      if (Cb) Cb[(size_t)grow * ldcb + gcol] = (__bf16)v2;
    }
}

// ---------------- flash MHA (non-causal), one wave per (b,h,qtile) ----------------
// Vt is V transposed: Vt[d][b*S+s], ld = Mrows
__global__ __launch_bounds__(32) void attn_mha_kernel(const __bf16* __restrict__ Q,
                                                      const __bf16* __restrict__ Kb,
                                                      const __bf16* __restrict__ Vt,
                                                      __bf16* __restrict__ cat) {
  __shared__ __bf16 Ps[16][40];
  int b = blockIdx.z, h = blockIdx.y, qt = blockIdx.x;
  int lane = threadIdx.x, half = lane >> 4, lq = lane & 15;
  size_t base = (size_t)b * Sm;
  const __bf16* qrow = Q + (base + qt * 16 + lq) * Dm + h * DHm;
  v16bf aq0 = ld_frag2(qrow + half * 8, qrow + 16 + half * 8);
  v16bf aq1 = ld_frag2(qrow + 32 + half * 8, qrow + 48 + half * 8);
  float mi[8], li[8];
  v8f o[4];
  #pragma unroll
  for (int r = 0; r < 8; ++r) { mi[r] = -1e30f; li[r] = 0.f; }
  #pragma unroll
  for (int nb = 0; nb < 4; ++nb)
    #pragma unroll
    for (int r = 0; r < 8; ++r) o[nb][r] = 0.f;

  for (int kv0 = 0; kv0 < Sm; kv0 += 32) {
    v8f s0, s1;
    #pragma unroll
    for (int nblk = 0; nblk < 2; ++nblk) {
      const __bf16* krow = Kb + (base + kv0 + nblk * 16 + lq) * Dm + h * DHm;
      v8f acc;
      #pragma unroll
      for (int r = 0; r < 8; ++r) acc[r] = 0.f;
      acc = wmma_bf16(aq0, ld_frag2(krow + half * 8, krow + 16 + half * 8), acc);
      acc = wmma_bf16(aq1, ld_frag2(krow + 32 + half * 8, krow + 48 + half * 8), acc);
      if (nblk == 0) s0 = acc; else s1 = acc;
    }
    float p0[8], p1[8], al[8];
    #pragma unroll
    for (int r = 0; r < 8; ++r) {
      s0[r] *= 0.125f; s1[r] *= 0.125f;
      float rm = fmaxf(s0[r], s1[r]);
      for (int off = 1; off < 16; off <<= 1) rm = fmaxf(rm, __shfl_xor(rm, off, 32));
      float mnew = fmaxf(mi[r], rm);
      al[r] = expf(mi[r] - mnew);
      p0[r] = expf(s0[r] - mnew);
      p1[r] = expf(s1[r] - mnew);
      float rs = p0[r] + p1[r];
      for (int off = 1; off < 16; off <<= 1) rs += __shfl_xor(rs, off, 32);
      li[r] = li[r] * al[r] + rs;
      mi[r] = mnew;
    }
    #pragma unroll
    for (int nb = 0; nb < 4; ++nb)
      #pragma unroll
      for (int r = 0; r < 8; ++r) o[nb][r] *= al[r];
    __syncthreads();
    #pragma unroll
    for (int r = 0; r < 8; ++r) {
      Ps[8 * half + r][lq]      = (__bf16)p0[r];
      Ps[8 * half + r][16 + lq] = (__bf16)p1[r];
    }
    __syncthreads();
    v16bf ap = ld_frag2(&Ps[lq][half * 8], &Ps[lq][16 + half * 8]);
    #pragma unroll
    for (int nb = 0; nb < 4; ++nb) {
      const __bf16* vp = Vt + (size_t)(h * DHm + nb * 16 + lq) * Mrows + base + kv0 + half * 8;
      o[nb] = wmma_bf16(ap, ld_frag2(vp, vp + 16), o[nb]);
    }
    __syncthreads();
  }
  #pragma unroll
  for (int nb = 0; nb < 4; ++nb)
    #pragma unroll
    for (int r = 0; r < 8; ++r) {
      size_t grow = base + qt * 16 + 8 * half + r;
      int gcol = Dm + h * DHm + nb * 16 + lq;  // ctx half of concat buffer
      cat[grow * (2 * Dm) + gcol] = (__bf16)(o[nb][r] / li[r]);
    }
}

// ------- causal mini-attention (MID=32) + alpha*transform + exact GELU -------
// Nt is normed transposed: Nt[d][b*S+s], ld = Mrows
__global__ __launch_bounds__(32) void attn2_kernel(const __bf16* __restrict__ qs,
                                                   const __bf16* __restrict__ ks,
                                                   const __bf16* __restrict__ Nt,
                                                   const float* __restrict__ transform,
                                                   const float* __restrict__ alpha_arr, int layer,
                                                   __bf16* __restrict__ inner) {
  __shared__ __bf16 Ps[16][40];
  int b = blockIdx.z, cbk = blockIdx.y, qt = blockIdx.x;
  int lane = threadIdx.x, half = lane >> 4, lq = lane & 15;
  size_t base = (size_t)b * Sm;
  const __bf16* qrow = qs + (base + qt * 16 + lq) * MIDm;
  v16bf aq = ld_frag2(qrow + half * 8, qrow + 16 + half * 8);
  float mi[8], li[8];
  v8f o[4];
  #pragma unroll
  for (int r = 0; r < 8; ++r) { mi[r] = -1e30f; li[r] = 0.f; }
  #pragma unroll
  for (int nb = 0; nb < 4; ++nb)
    #pragma unroll
    for (int r = 0; r < 8; ++r) o[nb][r] = 0.f;

  for (int kv0 = 0; kv0 <= qt * 16; kv0 += 32) {
    v8f s0, s1;
    #pragma unroll
    for (int nblk = 0; nblk < 2; ++nblk) {
      const __bf16* krow = ks + (base + kv0 + nblk * 16 + lq) * MIDm;
      v8f acc;
      #pragma unroll
      for (int r = 0; r < 8; ++r) acc[r] = 0.f;
      acc = wmma_bf16(aq, ld_frag2(krow + half * 8, krow + 16 + half * 8), acc);
      if (nblk == 0) s0 = acc; else s1 = acc;
    }
    float p0[8], p1[8], al[8];
    #pragma unroll
    for (int r = 0; r < 8; ++r) {
      int irow = qt * 16 + 8 * half + r;
      int j0 = kv0 + lq, j1 = kv0 + 16 + lq;
      s0[r] = (j0 > irow) ? -1e30f : s0[r] * 0.1767766953f;  // 1/sqrt(32)
      s1[r] = (j1 > irow) ? -1e30f : s1[r] * 0.1767766953f;
      float rm = fmaxf(s0[r], s1[r]);
      for (int off = 1; off < 16; off <<= 1) rm = fmaxf(rm, __shfl_xor(rm, off, 32));
      float mnew = fmaxf(mi[r], rm);
      al[r] = expf(mi[r] - mnew);
      p0[r] = expf(s0[r] - mnew);
      p1[r] = expf(s1[r] - mnew);
      float rs = p0[r] + p1[r];
      for (int off = 1; off < 16; off <<= 1) rs += __shfl_xor(rs, off, 32);
      li[r] = li[r] * al[r] + rs;
      mi[r] = mnew;
    }
    #pragma unroll
    for (int nb = 0; nb < 4; ++nb)
      #pragma unroll
      for (int r = 0; r < 8; ++r) o[nb][r] *= al[r];
    __syncthreads();
    #pragma unroll
    for (int r = 0; r < 8; ++r) {
      Ps[8 * half + r][lq]      = (__bf16)p0[r];
      Ps[8 * half + r][16 + lq] = (__bf16)p1[r];
    }
    __syncthreads();
    v16bf ap = ld_frag2(&Ps[lq][half * 8], &Ps[lq][16 + half * 8]);
    #pragma unroll
    for (int nb = 0; nb < 4; ++nb) {
      const __bf16* vp = Nt + (size_t)(cbk * 64 + nb * 16 + lq) * Mrows + base + kv0 + half * 8;
      o[nb] = wmma_bf16(ap, ld_frag2(vp, vp + 16), o[nb]);
    }
    __syncthreads();
  }
  float alpha = alpha_arr[layer];
  #pragma unroll
  for (int nb = 0; nb < 4; ++nb)
    #pragma unroll
    for (int r = 0; r < 8; ++r) {
      size_t grow = base + qt * 16 + 8 * half + r;
      int gcol = cbk * 64 + nb * 16 + lq;
      float v2 = o[nb][r] / li[r] + alpha * transform[grow * Dm + gcol];
      float ge = 0.5f * v2 * (1.0f + erff(v2 * 0.70710678118f));  // exact gelu
      inner[grow * Dm + gcol] = (__bf16)ge;
    }
}

// ---------------- host launcher ----------------
extern "C" void kernel_launch(void* const* d_in, const int* in_sizes, int n_in,
                              void* d_out, int out_size, void* d_ws, size_t ws_size,
                              hipStream_t stream) {
  (void)in_sizes; (void)n_in; (void)out_size; (void)ws_size;
  const int*   tokens = (const int*)d_in[0];
  const float* te     = (const float*)d_in[1];
  const float* pe     = (const float*)d_in[2];
  const float* rb     = (const float*)d_in[3];
  const float* cbas   = (const float*)d_in[4];
  const float* ln1g   = (const float*)d_in[5];
  const float* ln1b   = (const float*)d_in[6];
  const float* qw     = (const float*)d_in[7];
  const float* qbb    = (const float*)d_in[8];
  const float* kw     = (const float*)d_in[9];
  const float* kbb    = (const float*)d_in[10];
  const float* vw     = (const float*)d_in[11];
  const float* vbb    = (const float*)d_in[12];
  const float* sw     = (const float*)d_in[13];
  const float* sbb    = (const float*)d_in[14];
  const float* recipe = (const float*)d_in[15];
  const float* ow     = (const float*)d_in[16];
  const float* obb    = (const float*)d_in[17];
  const float* alphap = (const float*)d_in[18];
  const float* lnfg   = (const float*)d_in[19];
  const float* lnfb   = (const float*)d_in[20];
  float* logits = (float*)d_out;

  char* p = (char*)d_ws;
  auto alloc = [&](size_t bytes) -> char* {
    char* r = p; p += (bytes + 255) & ~(size_t)255; return r;
  };
  __bf16* qw_bf   = (__bf16*)alloc((size_t)Lm * Dm * Dm * 2);
  __bf16* kw_bf   = (__bf16*)alloc((size_t)Lm * Dm * Dm * 2);
  __bf16* vw_bf   = (__bf16*)alloc((size_t)Lm * Dm * Dm * 2);
  __bf16* ow_bf   = (__bf16*)alloc((size_t)Lm * Dm * Dm * 2);
  __bf16* sw_bf   = (__bf16*)alloc((size_t)Lm * 2 * Dm * Dm * 2);
  __bf16* tembT   = (__bf16*)alloc((size_t)Dm * Vv * 2);
  float*  x       = (float*) alloc((size_t)Mrows * Dm * 4);
  __bf16* cat     = (__bf16*)alloc((size_t)Mrows * 2 * Dm * 2);  // [normed | ctx]
  __bf16* qbuf    = (__bf16*)alloc((size_t)Mrows * Dm * 2);
  __bf16* kbuf    = (__bf16*)alloc((size_t)Mrows * Dm * 2);
  __bf16* vbuf    = (__bf16*)alloc((size_t)Mrows * Dm * 2);
  __bf16* vT      = (__bf16*)alloc((size_t)Dm * Mrows * 2);
  __bf16* nT      = (__bf16*)alloc((size_t)Dm * Mrows * 2);
  __bf16* query_b = (__bf16*)alloc((size_t)Mrows * Dm * 2);
  float*  scores  = (float*) alloc((size_t)Mrows * NNm * 4);
  float*  rec_sm  = (float*) alloc((size_t)NNm * NCOMBm * 4);
  float*  basis   = (float*) alloc((size_t)NCOMBm * Dm * 4);
  __bf16* neuronT = (__bf16*)alloc((size_t)Dm * NNm * 2);
  float*  row_sum = (float*) alloc((size_t)NRm * MIDm * 4);
  float*  col_s_j = (float*) alloc((size_t)NCm * MIDm * 4);
  float*  col_s_i = (float*) alloc((size_t)NCm * Dm * 4);
  __bf16* qs_bf   = (__bf16*)alloc((size_t)Mrows * MIDm * 2);
  __bf16* ks_bf   = (__bf16*)alloc((size_t)Mrows * MIDm * 2);
  float*  trans   = (float*) alloc((size_t)Mrows * Dm * 4);
  __bf16* inner   = (__bf16*)alloc((size_t)Mrows * Dm * 2);
  __bf16* xf_bf   = (__bf16*)alloc((size_t)Mrows * Dm * 2);

  auto cvt = [&](const float* src, __bf16* dst, long long n) {
    cvt_bf16_kernel<<<(unsigned)((n + 255) / 256), 256, 0, stream>>>(src, dst, n);
  };
  cvt(qw, qw_bf, (long long)Lm * Dm * Dm);
  cvt(kw, kw_bf, (long long)Lm * Dm * Dm);
  cvt(vw, vw_bf, (long long)Lm * Dm * Dm);
  cvt(ow, ow_bf, (long long)Lm * Dm * Dm);
  cvt(sw, sw_bf, (long long)Lm * 2 * Dm * Dm);
  {
    long long n = (long long)Vv * Dm;
    transpose_cvt_kernel<<<(unsigned)((n + 255) / 256), 256, 0, stream>>>(te, tembT);
  }
  basis_kernel<<<NCOMBm, 128, 0, stream>>>(rb, cbas, basis);
  sums_kernel<<<NRm, 256, 0, stream>>>(rb, cbas, row_sum, col_s_j, col_s_i);
  embed_kernel<<<Mrows, 128, 0, stream>>>(tokens, te, pe, x);

  dim3 g512(Dm / 64, Mrows / 64);                 // (8, 64)
  dim3 gtr(Dm / 32, Mrows / 32);                  // (16, 128)
  for (int l = 0; l < Lm; ++l) {
    ln_kernel<<<Mrows, 128, 0, stream>>>(x, ln1g + l * Dm, ln1b + l * Dm, cat, 2 * Dm);
    // QKV projections (bf16 out)
    wmma_gemm_kernel<<<g512, 128, 0, stream>>>(cat, 2 * Dm, qw_bf + (size_t)l * Dm * Dm, Dm,
        qbb + l * Dm, nullptr, 0, nullptr, 0, qbuf, Dm, Dm);
    wmma_gemm_kernel<<<g512, 128, 0, stream>>>(cat, 2 * Dm, kw_bf + (size_t)l * Dm * Dm, Dm,
        kbb + l * Dm, nullptr, 0, nullptr, 0, kbuf, Dm, Dm);
    wmma_gemm_kernel<<<g512, 128, 0, stream>>>(cat, 2 * Dm, vw_bf + (size_t)l * Dm * Dm, Dm,
        vbb + l * Dm, nullptr, 0, nullptr, 0, vbuf, Dm, Dm);
    // transposes for the P·V paths
    transpose_bf16_kernel<<<gtr, 256, 0, stream>>>(vbuf, Dm, vT, Mrows, Dm);
    transpose_bf16_kernel<<<gtr, 256, 0, stream>>>(cat, 2 * Dm, nT, Mrows, Dm);
    // full MHA -> ctx half of cat
    attn_mha_kernel<<<dim3(Sm / 16, Hm, Bm), 32, 0, stream>>>(qbuf, kbuf, vT, cat);
    // query = cat @ sw + sb
    wmma_gemm_kernel<<<g512, 128, 0, stream>>>(cat, 2 * Dm, sw_bf + (size_t)l * 2 * Dm * Dm, Dm,
        sbb + l * Dm, nullptr, 0, nullptr, 0, query_b, Dm, 2 * Dm);
    // recipe softmax + neuron embeddings (transposed bf16)
    recsm_kernel<<<NNm, 64, 0, stream>>>(recipe + (size_t)l * NNm * NCOMBm, rec_sm);
    neuron_kernel<<<NNm, 256, 0, stream>>>(rec_sm, basis, neuronT);
    // scores = query @ neuron_emb.T
    wmma_gemm_kernel<<<dim3(NNm / 64, Mrows / 64), 128, 0, stream>>>(query_b, Dm, neuronT, NNm,
        nullptr, nullptr, 0, scores, NNm, nullptr, 0, Dm);
    // top-k routing -> qs / ks2 / transform
    route_kernel<<<Mrows, 64, 0, stream>>>(scores, rec_sm, row_sum, col_s_j, col_s_i,
                                           qs_bf, ks_bf, trans);
    // causal mini-attention + gelu -> inner
    attn2_kernel<<<dim3(Sm / 16, Dm / 64, Bm), 32, 0, stream>>>(qs_bf, ks_bf, nT, trans,
                                                                alphap, l, inner);
    // x = x + inner @ ow + ob
    wmma_gemm_kernel<<<g512, 128, 0, stream>>>(inner, Dm, ow_bf + (size_t)l * Dm * Dm, Dm,
        obb + l * Dm, x, Dm, x, Dm, nullptr, 0, Dm);
  }
  ln_kernel<<<Mrows, 128, 0, stream>>>(x, lnfg, lnfb, xf_bf, Dm);
  // logits = xf @ token_emb.T  (4096 x 32000 x 512)
  wmma_gemm_kernel<<<dim3(Vv / 64, Mrows / 64), 128, 0, stream>>>(xf_bf, Dm, tembT, Vv,
      nullptr, nullptr, 0, logits, Vv, nullptr, 0, Dm);
}
